// Head_86921548136841
// MI455X (gfx1250) — matrix operand order repaired and checked by reading, not compiled, not run on hardware
//
#include <hip/hip_runtime.h>

typedef __bf16 bf16_t;
typedef __attribute__((ext_vector_type(16))) __bf16 v16bf;
typedef __attribute__((ext_vector_type(8)))  float  v8f;
typedef __attribute__((ext_vector_type(4)))  float  v4f;

#define E_DIM   1024
#define DH      32
#define M_FEAT  256
#define KSTEPS  32          // E_DIM / 32
#define TOK_PER_WAVE  16
#define WAVES_PER_BLK 8
#define TOK_PER_BLK   128
#define N_TOKENS      16384 // B*T = 8*2048

// LDS layout (bytes): B-fragments pre-swizzled to WMMA layout, bf16
#define WQ_OFF  0                 // 64 fragments * 1KB = 64KB
#define WK_OFF  65536             // 64KB
#define W2_OFF  131072            // 16 fragments * 1KB = 16KB
#define Z_OFF   147456            // 8 waves * 1KB z-scratch
#define LDS_BYTES (147456 + WAVES_PER_BLK * 1024)   // 155648

__global__ __launch_bounds__(256)
void head_favor_kernel(const float* __restrict__ x,
                       const float* __restrict__ Wq, const float* __restrict__ bq,
                       const float* __restrict__ Wk, const float* __restrict__ bk,
                       const float* __restrict__ w,
                       float* __restrict__ out)
{
    extern __shared__ char smem[];

    const int tid   = threadIdx.x;
    const int wave  = tid >> 5;
    const int lane  = tid & 31;
    const int lrow  = lane & 15;   // row (A) / col (B,C) index within 16-group
    const int khalf = lane >> 4;   // which K-half this lane holds

    // ---------------- stage Wq / Wk / w into LDS as bf16 WMMA B-fragments --------------
    // B-fragment lane layout (16-bit, 32xK16): lane L holds N = L&15,
    // elements i=0..7 -> K = (L>>4)*8 + i ; i=8..15 -> K = 16 + (L>>4)*8 + (i-8).
    for (int id = tid; id < 4608; id += 256) {
        union { uint4 u[2]; bf16_t h[16]; } frag;
        const int L = id & 31;
        if (id < 4096) {
            // Wq fragments: ids 0..2047, Wk: 2048..4095.  fragment = kb*2 + h (64 per matrix)
            const float* Wsrc = (id < 2048) ? Wq : Wk;
            const int fh = (id >> 5) & 63;           // kb*2 + h
            const int h  = fh & 1;
            const int kb = fh >> 1;
            const int N  = h * 16 + (L & 15);        // output column 0..31
            const int kbase = kb * 32 + (L >> 4) * 8;
            #pragma unroll
            for (int i = 0; i < 16; ++i)
                frag.h[i] = (bf16_t)Wsrc[(kbase + i + (i >= 8 ? 8 : 0)) * DH + N];
            char* dst = smem + ((id < 2048) ? WQ_OFF : WK_OFF) + fh * 1024 + L * 32;
            *(uint4*)dst        = frag.u[0];
            *(uint4*)(dst + 16) = frag.u[1];
        } else {
            // feature matrix w [M_FEAT][DH]: B2[k][n] = w[n][k]  (16 n-tiles)
            const int t2 = id - 4096;                // 0..511
            const int nt = t2 >> 5;
            const int N  = nt * 16 + (L & 15);
            const int kbase = (L >> 4) * 8;
            #pragma unroll
            for (int i = 0; i < 16; ++i)
                frag.h[i] = (bf16_t)w[N * DH + kbase + i + (i >= 8 ? 8 : 0)];
            char* dst = smem + W2_OFF + nt * 1024 + L * 32;
            *(uint4*)dst        = frag.u[0];
            *(uint4*)(dst + 16) = frag.u[1];
        }
    }
    __syncthreads();

    // ---------------- projection GEMMs: q = x*Wq, k = x*Wk (bf16 WMMA, f32 acc) --------
    const int tileBase = blockIdx.x * TOK_PER_BLK + wave * TOK_PER_WAVE;
    const float* xrow = x + (size_t)(tileBase + lrow) * E_DIM;

    v8f cq0 = {}, cq1 = {}, ck0 = {}, ck1 = {};

    const char* wqBase = smem + WQ_OFF + lane * 32;
    const char* wkBase = smem + WK_OFF + lane * 32;

    for (int kb = 0; kb < KSTEPS; ++kb) {
        // A-fragment (16-bit A 16x32 layout): lane holds row lrow,
        // i=0..7  -> K = khalf*8 + i ; i=8..15 -> K = 16 + khalf*8 + (i-8)
        const float* p = xrow + kb * 32 + khalf * 8;
        v4f f0 = *(const v4f*)(p);
        v4f f1 = *(const v4f*)(p + 4);
        v4f f2 = *(const v4f*)(p + 16);
        v4f f3 = *(const v4f*)(p + 20);
        union { v16bf v; bf16_t h[16]; } a;
        #pragma unroll
        for (int i = 0; i < 4; ++i) {
            a.h[i]      = (bf16_t)f0[i];
            a.h[4 + i]  = (bf16_t)f1[i];
            a.h[8 + i]  = (bf16_t)f2[i];
            a.h[12 + i] = (bf16_t)f3[i];
        }
        union { v16bf v; uint4 u[2]; } fq0, fq1, fk0, fk1;
        const char* q0p = wqBase + (kb * 2 + 0) * 1024;
        const char* q1p = wqBase + (kb * 2 + 1) * 1024;
        const char* k0p = wkBase + (kb * 2 + 0) * 1024;
        const char* k1p = wkBase + (kb * 2 + 1) * 1024;
        fq0.u[0] = *(const uint4*)q0p; fq0.u[1] = *(const uint4*)(q0p + 16);
        fq1.u[0] = *(const uint4*)q1p; fq1.u[1] = *(const uint4*)(q1p + 16);
        fk0.u[0] = *(const uint4*)k0p; fk0.u[1] = *(const uint4*)(k0p + 16);
        fk1.u[0] = *(const uint4*)k1p; fk1.u[1] = *(const uint4*)(k1p + 16);

        cq0 = __builtin_amdgcn_wmma_f32_16x16x32_bf16(false, a.v, false, fq0.v, (short)0, cq0, false, false);
        cq1 = __builtin_amdgcn_wmma_f32_16x16x32_bf16(false, a.v, false, fq1.v, (short)0, cq1, false, false);
        ck0 = __builtin_amdgcn_wmma_f32_16x16x32_bf16(false, a.v, false, fk0.v, (short)0, ck0, false, false);
        ck1 = __builtin_amdgcn_wmma_f32_16x16x32_bf16(false, a.v, false, fk1.v, (short)0, ck1, false, false);
    }

    // ---------------- bias, zsqr reduction, z -> LDS (A-layout for GEMM2) --------------
    const float bqv0 = bq[lrow], bqv1 = bq[16 + lrow];
    const float bkv0 = bk[lrow], bkv1 = bk[16 + lrow];

    float nh[8];   // -0.5 * zsqr for token (v + 8*khalf)
    bf16_t* zt = (bf16_t*)(smem + Z_OFF + wave * 1024);   // [16 rows][32 cols] bf16
    #pragma unroll
    for (int v = 0; v < 8; ++v) {
        float a0 = cq0[v] + bqv0, a1 = cq1[v] + bqv1;
        float c0 = ck0[v] + bkv0, c1 = ck1[v] + bkv1;
        float s = a0 * a0 + a1 * a1 + c0 * c0 + c1 * c1;
        s += __shfl_xor(s, 1, 32);
        s += __shfl_xor(s, 2, 32);
        s += __shfl_xor(s, 4, 32);
        s += __shfl_xor(s, 8, 32);     // sum over the 16 lanes of this half
        nh[v] = -0.5f * s;
        const int row = v + 8 * khalf;  // C-layout: M = vgpr + 8*(lane>>4)
        zt[row * 32 + lrow]      = (bf16_t)(a0 + c0);
        zt[row * 32 + 16 + lrow] = (bf16_t)(a1 + c1);
    }
    __builtin_amdgcn_wave_barrier();   // per-wave scratch; LDS pipe is in-order per wave

    // re-load z in A-fragment layout: row = lrow, chunks at cols khalf*8 and 16+khalf*8
    union { v16bf v; uint4 u[2]; } a2;
    const char* zl = smem + Z_OFF + wave * 1024 + lrow * 64 + khalf * 16;
    a2.u[0] = *(const uint4*)zl;
    a2.u[1] = *(const uint4*)(zl + 32);

    // ---------------- feature GEMM + stable A*cosh, write out ---------------------------
    float* outBase = out + (size_t)tileBase * M_FEAT;
    const char* w2Base = smem + W2_OFF + lane * 32;
    for (int nt = 0; nt < 16; ++nt) {
        union { v16bf v; uint4 u[2]; } b2;
        const char* p2 = w2Base + nt * 1024;
        b2.u[0] = *(const uint4*)p2;
        b2.u[1] = *(const uint4*)(p2 + 16);
        v8f d = {};
        d = __builtin_amdgcn_wmma_f32_16x16x32_bf16(false, a2.v, false, b2.v, (short)0, d, false, false);
        #pragma unroll
        for (int v = 0; v < 8; ++v) {
            const float dd = d[v];
            // A*cosh(d) = 0.5*(exp(d - 0.5 zsqr) + exp(-d - 0.5 zsqr))  -- overflow-safe
            const float r = 0.5f * (__expf(dd + nh[v]) + __expf(nh[v] - dd));
            outBase[(size_t)(v + 8 * khalf) * M_FEAT + nt * 16 + lrow] = r;
        }
    }
}

extern "C" void kernel_launch(void* const* d_in, const int* in_sizes, int n_in,
                              void* d_out, int out_size, void* d_ws, size_t ws_size,
                              hipStream_t stream) {
    (void)in_sizes; (void)n_in; (void)out_size; (void)d_ws; (void)ws_size;
    const float* x  = (const float*)d_in[0];
    const float* Wq = (const float*)d_in[1];
    const float* bq = (const float*)d_in[2];
    const float* Wk = (const float*)d_in[3];
    const float* bk = (const float*)d_in[4];
    // d_in[5] = Wv, d_in[6] = bv : dead in the reference, never touched
    const float* w  = (const float*)d_in[7];

    // allow >64KB dynamic LDS (CDNA5 WGP has 320KB); idempotent, capture-safe
    static bool attr_set = false;
    (void)attr_set;
    hipFuncSetAttribute((const void*)head_favor_kernel,
                        hipFuncAttributeMaxDynamicSharedMemorySize, LDS_BYTES);

    dim3 grid(N_TOKENS / TOK_PER_BLK);   // 128 blocks
    dim3 block(256);                     // 8 wave32 waves
    head_favor_kernel<<<grid, block, LDS_BYTES, stream>>>(
        x, Wq, bq, Wk, bk, w, (float*)d_out);
}
